// CrossModalAttention_71803263254918
// MI455X (gfx1250) — matrix operand order repaired and checked by reading, not compiled, not run on hardware
//
#include <hip/hip_runtime.h>
#include <hip/hip_bf16.h>
#include <math.h>

// ---- CDNA5 (gfx1250) wave32 WMMA types ----
typedef __attribute__((ext_vector_type(16))) __bf16 v16bf;
typedef __attribute__((ext_vector_type(8)))  __bf16 v8bf;
typedef __attribute__((ext_vector_type(4)))  __bf16 v4bf;
typedef __attribute__((ext_vector_type(8)))  float  v8f;
typedef __attribute__((ext_vector_type(4)))  unsigned int u32x4;
typedef __attribute__((ext_vector_type(8)))  int          i32x8;
typedef __attribute__((ext_vector_type(4)))  int          i32x4;

#define B_  16
#define C_  512
#define N_  1024
#define D_  64
#define WD_ 1e-5f

// TDM builtin: ROCm 7.2 (clang-22) 5-arg form. The therock toolchain ships the
// 6-arg form together with amd_gfx1250_TDM.h -- use that header's presence as
// the discriminator and fall back to manual LDS staging there / on host.
#if defined(__HIP_DEVICE_COMPILE__) && defined(__has_builtin)
#if __has_builtin(__builtin_amdgcn_tensor_load_to_lds) && \
    !__has_include(<hip/amd_detail/amd_gfx1250_TDM.h>)
#define USE_TDM 1
#endif
#endif
#ifndef USE_TDM
#define USE_TDM 0
#endif

static __device__ __forceinline__ v8f wmma_bf16(v16bf a, v16bf b, v8f c) {
  // D(16x16 f32) = A(16x32 bf16) * B(32x16 bf16) + C
  return __builtin_amdgcn_wmma_f32_16x16x32_bf16(
      /*neg_a=*/false, a, /*neg_b=*/false, b,
      /*c_mod=*/(short)0, c, /*reuse_a=*/false, /*reuse_b=*/false);
}

static __device__ __forceinline__ __bf16 f2bf(float f) { return (__bf16)f; }

static __device__ __forceinline__ v16bf cat8(v8bf lo, v8bf hi) {
  return __builtin_shufflevector(lo, hi, 0,1,2,3,4,5,6,7,8,9,10,11,12,13,14,15);
}

// A-tile (16x32 bf16): lane m holds K = kh*16+{0..15} contiguous
static __device__ __forceinline__ v16bf load_A_bf(const __bf16* __restrict__ row,
                                                  int k0, int kh) {
  v8bf lo = *(const v8bf*)(row + k0 + kh * 16);       // 16B aligned
  v8bf hi = *(const v8bf*)(row + k0 + kh * 16 + 8);   // 16B aligned
  return cat8(lo, hi);
}

// B-tile (32x16 bf16): lane m = column, K = kh*16+{0..15} contiguous
static __device__ __forceinline__ v16bf load_B_bf(const __bf16* __restrict__ row,
                                                  int k0, int kh) {
  v8bf lo = *(const v8bf*)(row + k0 + kh * 16);
  v8bf hi = *(const v8bf*)(row + k0 + kh * 16 + 8);
  return cat8(lo, hi);
}

// B-tile sourced from f32 weights (converted inline).
static __device__ __forceinline__ v16bf load_B_f32(const float* __restrict__ row,
                                                   int k0, int kh) {
  const float4 f0 = *(const float4*)(row + k0 + kh * 16);
  const float4 f1 = *(const float4*)(row + k0 + kh * 16 + 4);
  const float4 f2 = *(const float4*)(row + k0 + kh * 16 + 8);
  const float4 f3 = *(const float4*)(row + k0 + kh * 16 + 12);
  v16bf b;
  b[0]=f2bf(f0.x); b[1]=f2bf(f0.y); b[2]=f2bf(f0.z); b[3]=f2bf(f0.w);
  b[4]=f2bf(f1.x); b[5]=f2bf(f1.y); b[6]=f2bf(f1.z); b[7]=f2bf(f1.w);
  b[8]=f2bf(f2.x); b[9]=f2bf(f2.y); b[10]=f2bf(f2.z); b[11]=f2bf(f2.w);
  b[12]=f2bf(f3.x); b[13]=f2bf(f3.y); b[14]=f2bf(f3.z); b[15]=f2bf(f3.w);
  return b;
}

// -------------------------------------------------------------------------
// Kernel 0: scalar L2 penalty
// -------------------------------------------------------------------------
__global__ void l2_kernel(const float* __restrict__ qw, const float* __restrict__ qb,
                          const float* __restrict__ kw, const float* __restrict__ kb,
                          const float* __restrict__ vw, const float* __restrict__ vb,
                          const float* __restrict__ g,  float* __restrict__ out) {
  __shared__ float red[256];
  const float* arrs[7] = {qw, qb, kw, kb, vw, vb, g};
  const int    lens[7] = {D_*C_, D_, D_*C_, D_, C_*C_, C_, 1};
  float total = 0.0f;
  for (int a = 0; a < 7; ++a) {
    float s = 0.0f;
    for (int i = threadIdx.x; i < lens[a]; i += 256) {
      float t = arrs[a][i];
      s += t * t;
    }
    red[threadIdx.x] = s;
    __syncthreads();
    for (int st = 128; st > 0; st >>= 1) {
      if ((int)threadIdx.x < st) red[threadIdx.x] += red[threadIdx.x + st];
      __syncthreads();
    }
    if (threadIdx.x == 0) total += sqrtf(red[0]);
    __syncthreads();
  }
  if (threadIdx.x == 0) *out = WD_ * total;
}

// -------------------------------------------------------------------------
// Kernel 1: 1x1-conv projection as GEMM, 16x64 output strip per wave.
//   dst(n,o) = sum_c W[o,c] * src[b,c,n] + bias[o]
//   OUT_ON == 0 : dst stored (B, N, O)   (Q, K)
//   OUT_ON == 1 : dst stored (B, O, N)   (V, channel-major, i-contiguous stores)
// -------------------------------------------------------------------------
template <int OUT_ON>
__global__ void proj_kernel(const float* __restrict__ src,   // (B, C_, N_)
                            const float* __restrict__ W,     // (O, C_)
                            const float* __restrict__ bias,  // (O)
                            __bf16* __restrict__ dst, int O) {
  const int lane = threadIdx.x & 31;
  const int wave = threadIdx.x >> 5;
  const int m    = lane & 15;
  const int kh   = lane >> 4;

  const int ogroups = O >> 6;                 // 64-wide output strips
  const int tilesN  = N_ >> 4;
  const int tpb     = tilesN * ogroups;
  const int tid = blockIdx.x * 8 + wave;
  const int b   = tid / tpb;
  const int r   = tid % tpb;
  const int n0  = (r / ogroups) << 4;
  const int o0  = (r % ogroups) << 6;

  const float* sb = src + (size_t)b * C_ * N_;
  const float* ap = sb + n0 + m;              // A(mm,kk) = sb[kk*N_ + n0+mm]
  const float* wrow[4];
#pragma unroll
  for (int oo = 0; oo < 4; ++oo)
    wrow[oo] = W + (size_t)(o0 + oo * 16 + m) * C_;

  v8f acc[4] = {};
  for (int k0 = 0; k0 < C_; k0 += 32) {
    __builtin_prefetch(ap + (size_t)(k0 + 32) * N_, 0, 1);
    v16bf a;
#pragma unroll
    for (int t = 0; t < 8; ++t)
      a[t] = f2bf(ap[(size_t)(k0 + kh * 16 + t) * N_]);
#pragma unroll
    for (int t = 0; t < 8; ++t)
      a[8 + t] = f2bf(ap[(size_t)(k0 + kh * 16 + 8 + t) * N_]);
#pragma unroll
    for (int oo = 0; oo < 4; ++oo)
      acc[oo] = wmma_bf16(a, load_B_f32(wrow[oo], k0, kh), acc[oo]);
  }

#pragma unroll
  for (int oo = 0; oo < 4; ++oo) {
    const int ocol = o0 + oo * 16 + m;        // N-dim of tile = o index
    const float bv = bias[ocol];
    if (OUT_ON) {
      // (B, O, N): per lane 8 consecutive n -> one 16B bf16 store
      v8bf pk;
#pragma unroll
      for (int rr = 0; rr < 8; ++rr) pk[rr] = f2bf(acc[oo][rr] + bv);
      *(v8bf*)(dst + (size_t)b * O * N_ + (size_t)ocol * N_ + n0 + kh * 8) = pk;
    } else {
      // (B, N, O): strided b16 stores
#pragma unroll
      for (int rr = 0; rr < 8; ++rr) {
        const int mm = rr + kh * 8;
        dst[(size_t)b * N_ * O + (size_t)(n0 + mm) * O + ocol] =
            f2bf(acc[oo][rr] + bv);
      }
    }
  }
}

// -------------------------------------------------------------------------
// Kernel 2: energy[b,i,j] = sum_d Q[b,i,d] * K[b,j,d]; 16x64 strip per wave.
// -------------------------------------------------------------------------
__global__ void energy_kernel(const __bf16* __restrict__ Q,   // (B,N,D) bf16
                              const __bf16* __restrict__ K,   // (B,N,D) bf16
                              float* __restrict__ E) {        // (B,N,N) f32
  const int lane = threadIdx.x & 31;
  const int wave = threadIdx.x >> 5;
  const int m    = lane & 15;
  const int kh   = lane >> 4;

  const int tid = blockIdx.x * 8 + wave;      // 16 b * 64 i-tiles * 16 j-groups
  const int b   = tid >> 10;
  const int r   = tid & 1023;
  const int i0  = (r >> 4) << 4;
  const int j0  = (r & 15) << 6;

  const __bf16* qrow = Q + (size_t)b * N_ * D_ + (size_t)(i0 + m) * D_;
  const __bf16* krow[4];
#pragma unroll
  for (int jj = 0; jj < 4; ++jj)
    krow[jj] = K + (size_t)b * N_ * D_ + (size_t)(j0 + jj * 16 + m) * D_;

  v8f acc[4] = {};
#pragma unroll
  for (int k0 = 0; k0 < D_; k0 += 32) {
    const v16bf a = load_A_bf(qrow, k0, kh);
#pragma unroll
    for (int jj = 0; jj < 4; ++jj)
      acc[jj] = wmma_bf16(a, load_B_bf(krow[jj], k0, kh), acc[jj]);
  }

  float* ep = E + (size_t)b * N_ * N_;
#pragma unroll
  for (int jj = 0; jj < 4; ++jj)
#pragma unroll
    for (int rr = 0; rr < 8; ++rr) {
      const int mm = rr + kh * 8;
      ep[(size_t)(i0 + mm) * N_ + j0 + jj * 16 + m] = acc[jj][rr];
    }
}

// -------------------------------------------------------------------------
// Kernel 3: row softmax (1024 wide), f32 in -> bf16 out IN PLACE.
// Each wave owns one row: all f32 loads complete (values consumed by exp)
// before the bf16 stores issue, so the in-row aliasing is race-free.
// bf16 row occupies the first 2048B of the 4096B row slot.
// -------------------------------------------------------------------------
__global__ void softmax_kernel(float* __restrict__ E) {
  const int lane = threadIdx.x & 31;
  const int wave = threadIdx.x >> 5;
  const size_t row = (size_t)blockIdx.x * 8 + wave;   // 16*1024 rows
  float4* p = (float4*)(E + row * N_);                // 256 float4 per row
  __bf16* bfrow = (__bf16*)(E + row * N_);            // bf16 output, same slot

  float4 v[8];
  float mx = -3.0e38f;
#pragma unroll
  for (int t = 0; t < 8; ++t) {
    v[t] = p[lane + 32 * t];
    mx = fmaxf(mx, fmaxf(fmaxf(v[t].x, v[t].y), fmaxf(v[t].z, v[t].w)));
  }
#pragma unroll
  for (int off = 16; off > 0; off >>= 1) mx = fmaxf(mx, __shfl_xor(mx, off, 32));

  float s = 0.0f;
#pragma unroll
  for (int t = 0; t < 8; ++t) {
    v[t].x = __expf(v[t].x - mx); v[t].y = __expf(v[t].y - mx);
    v[t].z = __expf(v[t].z - mx); v[t].w = __expf(v[t].w - mx);
    s += v[t].x + v[t].y + v[t].z + v[t].w;
  }
#pragma unroll
  for (int off = 16; off > 0; off >>= 1) s += __shfl_xor(s, off, 32);

  const float inv = 1.0f / s;
  asm volatile("" ::: "memory");   // keep stores after all row loads
#pragma unroll
  for (int t = 0; t < 8; ++t) {
    v4bf pk;
    pk[0] = f2bf(v[t].x * inv); pk[1] = f2bf(v[t].y * inv);
    pk[2] = f2bf(v[t].z * inv); pk[3] = f2bf(v[t].w * inv);
    *(v4bf*)(bfrow + 4 * lane + 128 * t) = pk;       // 8B aligned
  }
}

// -------------------------------------------------------------------------
// Kernel 4: out[b,c,i] = gamma * sum_j attn[b,i,j] * V[b,c,j] + x[b,c,i] + l2
// Block = 8 waves = 128x64 (i x c) macro-tile. V k-slices (64 rows x 32 cols
// bf16, row stride N) are staged into LDS, double-buffered.
//  - TDM path: one TENSOR_LOAD_TO_LDS per slice, issued by wave 0, tracked
//    with TENSORcnt. D# pad fields (interval=16 DWORDs, amount=4 DWORDs)
//    give an 80B LDS row pitch -> conflict-free 16-lane fragment reads.
//  - Fallback: cooperative global_load_b128 + ds_store_b128.
// attn rows are bf16 (2048B payload in a 4096B row slot).
// -------------------------------------------------------------------------
#define VSTRIDE 80   // 64B of k-slice data + 16B pad per channel row

#if USE_TDM
static __device__ __forceinline__ void tdm_load_slice(unsigned ldsoff,
                                                      const __bf16* gsrc) {
  const unsigned long long ga = (unsigned long long)(size_t)gsrc;
  u32x4 g0;
  g0[0] = 1u;                                        // count=1, user mode
  g0[1] = ldsoff;                                    // lds_addr (bytes)
  g0[2] = (unsigned)(ga & 0xFFFFFFFFu);              // global_addr[31:0]
  g0[3] = (unsigned)((ga >> 32) & 0x01FFFFFFu)       // global_addr[56:32]
        | (2u << 30);                                // type = 2 ("image")
  i32x8 g1;
  g1[0] = (int)((1u << 16)      // data_size = 1 -> 2-byte elements
              | (1u << 20)      // pad_enable
              | (3u << 22)      // pad_interval: 16 DWORDs (64B)
              | (3u << 25));    // pad_amount:   4 DWORDs (16B) -> 80B pitch
  g1[1] = (int)(((unsigned)N_ & 0xFFFFu) << 16);     // tensor_dim0 lo16 (1024)
  g1[2] = (int)((((unsigned)N_ >> 16) & 0xFFFFu)     // tensor_dim0 hi16
              | (((unsigned)C_ & 0xFFFFu) << 16));   // tensor_dim1 lo16 (512)
  g1[3] = (int)((((unsigned)C_ >> 16) & 0xFFFFu)     // tensor_dim1 hi16
              | (32u << 16));                        // tile_dim0 = 32
  g1[4] = 64;                                        // tile_dim1 = 64, tile_dim2 = 0
  g1[5] = N_;                                        // tensor_dim0_stride lo32
  g1[6] = 0;
  g1[7] = 0;
  const i32x4 z = {0, 0, 0, 0};                      // dims 2+ unused
  __builtin_amdgcn_tensor_load_to_lds(g0, g1, z, z, /*cpol=*/0);
}
#endif

__global__ void out_kernel(const char*   __restrict__ attn,   // bf16 rows, 4096B stride
                           const __bf16* __restrict__ V,      // (B,C,N) bf16
                           const float*  __restrict__ x,      // (B,C,N) f32
                           const float*  __restrict__ gamma,
                           const float*  __restrict__ l2p,
                           float* __restrict__ out) {         // (B,C,N) f32
  __shared__ __align__(16) unsigned char vsm[2][64 * VSTRIDE];

  const int lane = threadIdx.x & 31;
  const int wave = threadIdx.x >> 5;
  const int m    = lane & 15;
  const int kh   = lane >> 4;

  const int b  = blockIdx.x >> 6;             // 16 b * 8 ig * 8 cg = 1024 blocks
  const int r  = blockIdx.x & 63;
  const int i0 = (r >> 3) << 7;               // 128-row macro tile
  const int c0 = (r & 7) << 6;                // 64-channel strip
  const int iw = i0 + wave * 16;              // this wave's 16 rows

  const __bf16* arow =
      (const __bf16*)(attn + ((size_t)b * N_ + iw + m) * 4096);
  const __bf16* strip = V + (size_t)b * C_ * N_ + (size_t)c0 * N_;

#if USE_TDM
  // Generic pointers into LDS: low 32 bits are the LDS byte offset.
  const unsigned lds0 = (unsigned)(size_t)&vsm[0][0];
  const unsigned lds1 = lds0 + 64 * VSTRIDE;
  if (wave == 0) {
    tdm_load_slice(lds0, strip);           // slice 0 -> buf 0
    tdm_load_slice(lds1, strip + 32);      // slice 1 -> buf 1
    __builtin_amdgcn_s_wait_tensorcnt(1);  // slice 0 resident (in-order)
  }
  __syncthreads();
#else
  // cooperative staging: thread -> (channel row, 16B chunk of the 64B k-slice)
  const int vr = threadIdx.x >> 2;            // 0..63
  const int vchunk = threadIdx.x & 3;         // 0..3
  const __bf16* vsrc = strip + (size_t)vr * N_ + vchunk * 8;
  unsigned char* vdst = &vsm[0][0] + vr * VSTRIDE + vchunk * 16;
  *(uint4*)(vdst) = *(const uint4*)(vsrc);                          // slice 0
  *(uint4*)(vdst + 64 * VSTRIDE) = *(const uint4*)(vsrc + 32);      // slice 1
  __syncthreads();
#endif

  v8f acc[4] = {};
  for (int i = 0; i < 32; ++i) {              // 32 k-slices of 32
    const int k0 = i << 5;
    const int cur = i & 1;

    __builtin_prefetch(arow + k0 + 32, 0, 1);
    const v16bf a = load_A_bf(arow, k0, kh);
#pragma unroll
    for (int cc = 0; cc < 4; ++cc) {
      const unsigned char* pb =
          &vsm[cur][0] + (cc * 16 + m) * VSTRIDE + kh * 32;
      v8bf lo = *(const v8bf*)(pb);
      v8bf hi = *(const v8bf*)(pb + 16);
      acc[cc] = wmma_bf16(a, cat8(lo, hi), acc[cc]);
    }
    __syncthreads();                          // all waves done reading buf[cur]

#if USE_TDM
    if (wave == 0) {
      if (i + 2 < 32)
        tdm_load_slice(cur ? lds1 : lds0, strip + k0 + 64);  // refill freed buf
      __builtin_amdgcn_s_wait_tensorcnt(1);   // slice i+1 resident
    }
    __syncthreads();
#else
    if (i + 2 < 32)
      *(uint4*)(vdst + cur * (64 * VSTRIDE)) = *(const uint4*)(vsrc + k0 + 64);
    __syncthreads();
#endif
  }

  const float g  = gamma[0];
  const float l2 = l2p[0];
  const size_t bbase = (size_t)b * C_ * N_;
#pragma unroll
  for (int cc = 0; cc < 4; ++cc) {
    const int c = c0 + cc * 16 + m;           // N-dim of tile = channel
    const size_t base = bbase + (size_t)c * N_ + iw + kh * 8; // 8 consecutive i
    const float4 x0 = *(const float4*)(x + base);
    const float4 x1 = *(const float4*)(x + base + 4);
    float4 o0v, o1v;
    o0v.x = g * acc[cc][0] + x0.x + l2;  o0v.y = g * acc[cc][1] + x0.y + l2;
    o0v.z = g * acc[cc][2] + x0.z + l2;  o0v.w = g * acc[cc][3] + x0.w + l2;
    o1v.x = g * acc[cc][4] + x1.x + l2;  o1v.y = g * acc[cc][5] + x1.y + l2;
    o1v.z = g * acc[cc][6] + x1.z + l2;  o1v.w = g * acc[cc][7] + x1.w + l2;
    *(float4*)(out + base)     = o0v;
    *(float4*)(out + base + 4) = o1v;
  }
}

// -------------------------------------------------------------------------
// Launch
// -------------------------------------------------------------------------
extern "C" void kernel_launch(void* const* d_in, const int* in_sizes, int n_in,
                              void* d_out, int out_size, void* d_ws, size_t ws_size,
                              hipStream_t stream) {
  (void)in_sizes; (void)n_in; (void)out_size; (void)ws_size;

  const float* x  = (const float*)d_in[0];
  const float* y  = (const float*)d_in[1];
  const float* qw = (const float*)d_in[2];
  const float* qb = (const float*)d_in[3];
  const float* kw = (const float*)d_in[4];
  const float* kb = (const float*)d_in[5];
  const float* vw = (const float*)d_in[6];
  const float* vb = (const float*)d_in[7];
  const float* gm = (const float*)d_in[8];
  float* out = (float*)d_out;

  // Workspace layout (bytes):
  //   [0,   2M)  Q  bf16 (B,N,D)
  //   [2M,  4M)  K  bf16 (B,N,D)
  //   [4M, 20M)  V  bf16 (B,C,N)
  //   [20M,84M)  E  f32  (B,N,N); softmax rewrites each 4096B row slot
  //              with its 2048B bf16 attn row in place
  //   [84M]      l2 scalar f32
  char* ws = (char*)d_ws;
  __bf16* Qbf = (__bf16*)(ws);
  __bf16* Kbf = (__bf16*)(ws + (size_t)2  * 1024 * 1024);
  __bf16* Vbf = (__bf16*)(ws + (size_t)4  * 1024 * 1024);
  float*  E   = (float*) (ws + (size_t)20 * 1024 * 1024);
  float*  l2s = (float*) (ws + (size_t)84 * 1024 * 1024);

  l2_kernel<<<1, 256, 0, stream>>>(qw, qb, kw, kb, vw, vb, gm, l2s);

  // Q/K: 16 b * 64 n-tiles * 1 o-group = 1024 waves -> 128 blocks
  proj_kernel<0><<<128, 256, 0, stream>>>(x, qw, qb, Qbf, D_);
  proj_kernel<0><<<128, 256, 0, stream>>>(y, kw, kb, Kbf, D_);
  // V: 16 b * 64 n-tiles * 8 o-groups = 8192 waves -> 1024 blocks
  proj_kernel<1><<<1024, 256, 0, stream>>>(y, vw, vb, Vbf, C_);

  // energy: 16 b * 64 i-tiles * 16 j-groups = 16384 waves -> 2048 blocks
  energy_kernel<<<2048, 256, 0, stream>>>(Qbf, Kbf, E);
  // softmax: 16384 rows, 8 rows per block (f32 -> bf16 in place)
  softmax_kernel<<<2048, 256, 0, stream>>>(E);
  // out: 16 b * 8 i-groups * 8 c-groups = 1024 blocks (128x64 macro tiles)
  out_kernel<<<1024, 256, 0, stream>>>((const char*)E, Vbf, x, gm, l2s, out);
}
